// PortfolioOptimizationLayer_79053168050594
// MI455X (gfx1250) — matrix operand order repaired and checked by reading, not compiled, not run on hardware
//
#include <hip/hip_runtime.h>

// MI455X (gfx1250, wave32) portfolio-QP solver.
// One wave32 per batch. Sigma is symmetrized into LDS as f16 once, then all
// 28 WMMA B-tiles (7 row tiles x 4 K-chunks) are hoisted into VGPRs (224
// VGPRs/lane; gfx1250 wave32 addresses up to 1024) so the 320-iteration loop
// does no Sigma reloads: each mat-vec is 28 x v_wmma_f32_16x16x32_f16 with
// only the w-vector round-tripping through LDS (7 b16 stores + 8 b128 loads).
// Simplex projection uses Michelot's exact active-set iteration (no sort).

typedef __attribute__((ext_vector_type(16))) _Float16 v16h;
typedef __attribute__((ext_vector_type(8)))  _Float16 v8h;
typedef __attribute__((ext_vector_type(8)))  float    v8f;

#define N_ASSETS   104
#define N_TILES    7      // rows padded to 112
#define K_CHUNKS   4      // K padded to 128
#define ROWSTRIDE  136    // f16 elems per row: 128 + 8 pad -> 272B, conflict-free
#define POWER_ITERS 20
#define PGD_ITERS   300
#define PROJ_ITERS  16

// sum across the 16 "owner" lanes (lanes 16..31 carry duplicates, so we only
// reduce within each 16-lane half; both halves end with the same total)
__device__ __forceinline__ float redsum16(float x) {
    x += __shfl_xor(x, 1);
    x += __shfl_xor(x, 2);
    x += __shfl_xor(x, 4);
    x += __shfl_xor(x, 8);
    return x;
}

// Load all Sigma B-tiles into registers (done once).
// B-matrix f16 32x16, contiguous k order chosen here (A matches): lanes 0-15 hold K=0..15, lanes 16-31 hold
// K=16..31 (consecutive), N = lane%16; row index = 16*t + N via symmetry.
__device__ __forceinline__ void load_B_tiles(const _Float16* __restrict__ sS,
                                             v16h Breg[N_TILES][K_CHUNKS],
                                             int lane) {
    const int jn  = lane & 15;
    const int kbB = (lane < 16) ? 0 : 16;
#pragma unroll
    for (int t = 0; t < N_TILES; ++t) {
        const _Float16* rowp = sS + (16 * t + jn) * ROWSTRIDE + kbB;
#pragma unroll
        for (int kc = 0; kc < K_CHUNKS; ++kc) {
            const _Float16* p = rowp + kc * 32;
            v8h lo = *(const v8h*)(p);      // K = kbB..kbB+7
            v8h hi = *(const v8h*)(p + 8);  // K = kbB+8..kbB+15
            Breg[t][kc] = __builtin_shufflevector(
                lo, hi, 0,1,2,3,4,5,6,7,8,9,10,11,12,13,14,15);
        }
    }
}

// y[t] = (Sigma * w)[16*t + (lane&15)] via WMMA; Sigma tiles register-resident.
// A-matrix f16 16x32, contiguous k order chosen here: lane<16 holds K 0..15, lane>=16 holds K 16..31 (matches B).
__device__ __forceinline__ void matvec_wmma(const v16h Breg[N_TILES][K_CHUNKS],
                                            const _Float16* __restrict__ sw,
                                            float* __restrict__ y, int lane) {
    const int kbA = (lane < 16) ? 0 : 16;

    v16h A[K_CHUNKS];
#pragma unroll
    for (int kc = 0; kc < K_CHUNKS; ++kc) {
        const _Float16* p = sw + kc * 32 + kbA;
        v8h lo = *(const v8h*)(p);        // K kbA..kbA+7
        v8h hi = *(const v8h*)(p + 8);    // K kbA+8..kbA+15
        A[kc] = __builtin_shufflevector(
            lo, hi, 0,1,2,3,4,5,6,7,8,9,10,11,12,13,14,15);
    }

#pragma unroll
    for (int t = 0; t < N_TILES; ++t) {
        v8f c = {0.f,0.f,0.f,0.f,0.f,0.f,0.f,0.f};
#pragma unroll
        for (int kc = 0; kc < K_CHUNKS; ++kc) {
            c = __builtin_amdgcn_wmma_f32_16x16x32_f16(
                    false, A[kc], false, Breg[t][kc], (short)0, c, false, false);
        }
        y[t] = c[0];  // D[m,n] identical over m: every lane holds y[16t + lane%16]
    }
}

__device__ __forceinline__ void store_vec(_Float16* __restrict__ sw,
                                          const float* __restrict__ w, int lane) {
    const int jn = lane & 15;
    if (lane < 16) {
#pragma unroll
        for (int t = 0; t < N_TILES; ++t)
            sw[16 * t + jn] = (_Float16)w[t];
    }
}

__global__ __launch_bounds__(32)
void pgd_portfolio_kernel(const float* __restrict__ g_returns,
                          const float* __restrict__ g_cov,
                          float* __restrict__ g_out) {
    __shared__ __align__(16) _Float16 sS[N_TILES * 16 * ROWSTRIDE]; // 112x136 f16
    __shared__ __align__(16) _Float16 sw[128];

    const int b    = blockIdx.x;
    const int lane = threadIdx.x;       // 0..31, one full wave32
    const int jn   = lane & 15;

    const float* C  = g_cov     + (size_t)b * N_ASSETS * N_ASSETS;
    const float* mu = g_returns + (size_t)b * N_ASSETS;

    // ---- zero LDS (padding rows/cols must be 0) ----
    {
        v8h z = {};
        const int nvec = (N_TILES * 16 * ROWSTRIDE) / 8;  // 1904 x 16B
        for (int k = lane; k < nvec; k += 32) ((v8h*)sS)[k] = z;
        for (int k = lane; k < 128 / 8; k += 32) ((v8h*)sw)[k] = z;
    }
    __syncthreads();

    // ---- Sigma = 0.5*(C + C^T) + 1e-6*I, as f16 in LDS ----
    for (int i = 0; i < N_ASSETS; ++i) {
        for (int jj = lane; jj < N_ASSETS; jj += 32) {
            float cij = C[i * N_ASSETS + jj];
            float cji = C[jj * N_ASSETS + i];   // gathered; hits L2 (43KB tile)
            float s = 0.5f * (cij + cji);
            if (i == jj) s += 1e-6f;
            sS[i * ROWSTRIDE + jj] = (_Float16)s;
        }
    }
    __syncthreads();

    // ---- hoist all Sigma WMMA B-tiles into registers (loop-invariant) ----
    v16h Breg[N_TILES][K_CHUNKS];
    load_B_tiles(sS, Breg, lane);

    // ---- per-lane element ownership: element e_t = 16*t + (lane&15) ----
    float valid[N_TILES], muv[N_TILES];
#pragma unroll
    for (int t = 0; t < N_TILES; ++t) {
        int e = 16 * t + jn;
        bool vv = (e < N_ASSETS);
        valid[t] = vv ? 1.f : 0.f;
        int ec = vv ? e : (N_ASSETS - 1);
        muv[t] = valid[t] * mu[ec];
    }

    float y[N_TILES];

    // ---- power iteration: lambda_max(Sigma) ----
    float v[N_TILES];
#pragma unroll
    for (int t = 0; t < N_TILES; ++t) v[t] = valid[t] * (1.f / N_ASSETS);
    store_vec(sw, v, lane);
    __syncthreads();

    for (int it = 0; it < POWER_ITERS; ++it) {
        matvec_wmma(Breg, sw, y, lane);
        float ss = 0.f;
#pragma unroll
        for (int t = 0; t < N_TILES; ++t) ss += y[t] * y[t]; // pad rows give y=0
        ss = redsum16(ss);
        float inv = 1.f / (sqrtf(ss) + 1e-12f);
#pragma unroll
        for (int t = 0; t < N_TILES; ++t) v[t] = y[t] * inv;
        __syncthreads();               // single-wave WG: compiler sched barrier
        store_vec(sw, v, lane);
        __syncthreads();
    }
    matvec_wmma(Breg, sw, y, lane);
    float lam = 0.f;
#pragma unroll
    for (int t = 0; t < N_TILES; ++t) lam += v[t] * y[t];
    lam = redsum16(lam);
    const float eta = 1.f / (2.2f * lam + 1e-8f);

    // ---- projected gradient descent ----
    float w[N_TILES];
#pragma unroll
    for (int t = 0; t < N_TILES; ++t) w[t] = valid[t] * (1.f / N_ASSETS);
    store_vec(sw, w, lane);
    __syncthreads();

    for (int it = 0; it < PGD_ITERS; ++it) {
        matvec_wmma(Breg, sw, y, lane);

        float z[N_TILES];
#pragma unroll
        for (int t = 0; t < N_TILES; ++t)
            z[t] = w[t] - eta * (2.f * y[t] - muv[t]);

        // exact simplex projection: Michelot active-set iteration
        float zs = 0.f;
#pragma unroll
        for (int t = 0; t < N_TILES; ++t) zs += valid[t] * z[t];
        float theta = (redsum16(zs) - 1.f) * (1.f / (float)N_ASSETS);
        for (int pi = 0; pi < PROJ_ITERS; ++pi) {
            float s = 0.f, cnt = 0.f;
#pragma unroll
            for (int t = 0; t < N_TILES; ++t) {
                bool a = (valid[t] != 0.f) && (z[t] > theta);
                s   += a ? z[t] : 0.f;
                cnt += a ? 1.f  : 0.f;
            }
            s   = redsum16(s);
            cnt = redsum16(cnt);
            theta = (s - 1.f) / fmaxf(cnt, 1.f);
        }
#pragma unroll
        for (int t = 0; t < N_TILES; ++t)
            w[t] = valid[t] * fmaxf(z[t] - theta, 0.f);

        __syncthreads();
        store_vec(sw, w, lane);
        __syncthreads();
    }

    // ---- output ----
    if (lane < 16) {
#pragma unroll
        for (int t = 0; t < N_TILES; ++t) {
            int e = 16 * t + jn;
            if (e < N_ASSETS) g_out[(size_t)b * N_ASSETS + e] = w[t];
        }
    }
}

extern "C" void kernel_launch(void* const* d_in, const int* in_sizes, int n_in,
                              void* d_out, int out_size, void* d_ws, size_t ws_size,
                              hipStream_t stream) {
    const float* g_returns = (const float*)d_in[0];
    const float* g_cov     = (const float*)d_in[1];
    float*       g_out     = (float*)d_out;
    const int B = in_sizes[0] / N_ASSETS;   // 4096
    pgd_portfolio_kernel<<<B, 32, 0, stream>>>(g_returns, g_cov, g_out);
}
